// BinaryConv2D_53334903881939
// MI455X (gfx1250) — 8-bit matrix operand order repaired and checked by reading, compiled, not run on hardware
//
#include <hip/hip_runtime.h>
#include <stdint.h>

typedef __attribute__((ext_vector_type(16))) int   v16i;
typedef __attribute__((ext_vector_type(8)))  float v8f;
typedef __attribute__((ext_vector_type(4)))  int   v4i;
typedef __attribute__((ext_vector_type(2)))  int   v2i;

#define NB   32
#define HH   56
#define WW   56
#define CIN  256
#define COUT 256
#define HP   58
#define WP   58
#define KDIM 2304
#define MTOT (NB*HH*WW)

#define FP8_P1 0x38
#define FP8_M1 0xB8

__global__ void prep_weights(const float* __restrict__ w,
                             uint8_t* __restrict__ binKT,
                             float* __restrict__ alpha) {
    __shared__ float red[256];
    const int co = blockIdx.x;
    const int t  = threadIdx.x;
    float s = 0.f;
    for (int k = t; k < KDIM; k += 256) {
        float v = w[(size_t)k * COUT + co];
        binKT[(size_t)co * KDIM + k] = (v > 0.f) ? FP8_P1 : FP8_M1;
        s += fabsf(v);
    }
    red[t] = s;
    __syncthreads();
    for (int off = 128; off > 0; off >>= 1) {
        if (t < off) red[t] += red[t + off];
        __syncthreads();
    }
    if (t == 0) alpha[co] = red[0] * (1.0f / KDIM);
}

__global__ void prep_input(const float* __restrict__ x,
                           uint8_t* __restrict__ binX,
                           float* __restrict__ beta) {
    __shared__ float red[256];
    const int p = blockIdx.x;
    const int t = threadIdx.x;
    const int n  = p / (HP * WP);
    const int r  = p % (HP * WP);
    const int hp = r / WP;
    const int wp = r % WP;
    const bool border = (hp == 0) | (hp == HP - 1) | (wp == 0) | (wp == WP - 1);
    float s = 0.f;
    uint8_t b = FP8_M1;
    if (!border) {
        float v = x[(size_t)(((n * HH) + (hp - 1)) * WW + (wp - 1)) * CIN + t];
        b = (v > 0.f) ? FP8_P1 : FP8_M1;
        s = fabsf(v);
    }
    binX[(size_t)p * CIN + t] = b;
    red[t] = s;
    __syncthreads();
    for (int off = 128; off > 0; off >>= 1) {
        if (t < off) red[t] += red[t + off];
        __syncthreads();
    }
    if (t == 0) beta[p] = red[0] * (1.0f / CIN);
}

__device__ __forceinline__ void async_copy64(uint32_t ldsOff, const uint8_t* g) {
    asm volatile(
        "global_load_async_to_lds_b128 %0, %1, off\n\t"
        "global_load_async_to_lds_b128 %0, %1, off offset:16\n\t"
        "global_load_async_to_lds_b128 %0, %1, off offset:32\n\t"
        "global_load_async_to_lds_b128 %0, %1, off offset:48"
        :: "v"(ldsOff), "v"(g) : "memory");
}

__device__ __forceinline__ void wait_async0() {
    asm volatile("s_wait_asynccnt 0x0" ::: "memory");
}
__device__ __forceinline__ void wait_ds0() {
    asm volatile("s_wait_dscnt 0x0" ::: "memory");
}

__global__ __launch_bounds__(256, 1)
void bconv_gemm(const uint8_t* __restrict__ binX,
                const uint8_t* __restrict__ binKT,
                const float*  __restrict__ beta,
                const float*  __restrict__ alpha,
                const float*  __restrict__ bias,
                float* __restrict__ out) {
    __shared__ uint8_t lA[2][128 * 128];
    __shared__ uint8_t lB[2][128 * 128];
    __shared__ float   kb[128];

    const int t    = threadIdx.x;
    const int lane = t & 31;
    const int wv   = t >> 5;
    const int wm   = wv >> 1;
    const int wn   = wv & 1;

    const int m0 = blockIdx.x * 128;
    const int n0 = blockIdx.y * 128;

    const int rA = t >> 1;
    const int pA = t & 1;
    const int mA   = m0 + rA;
    const int nImg = mA / (HH * WW);
    const int rmA  = mA % (HH * WW);
    const int hImg = rmA / WW;
    const int wImg = rmA % WW;
    const uint8_t* aBase = binX + (size_t)((nImg * HP + hImg) * WP + wImg) * CIN + pA * 64;

    const int cB = t >> 1;
    const int pB = t & 1;
    const uint8_t* bBase = binKT + (size_t)(n0 + cB) * KDIM + pB * 64;

    const uint32_t aDst0 = (uint32_t)(uintptr_t)&lA[0][rA * 128 + pA * 64];
    const uint32_t aDst1 = (uint32_t)(uintptr_t)&lA[1][rA * 128 + pA * 64];
    const uint32_t bDst0 = (uint32_t)(uintptr_t)&lB[0][cB * 128 + pB * 64];
    const uint32_t bDst1 = (uint32_t)(uintptr_t)&lB[1][cB * 128 + pB * 64];

    if (t < 128) {
        const int m  = m0 + t;
        const int n2 = m / (HH * WW);
        const int r2 = m % (HH * WW);
        const int h2 = r2 / WW;
        const int w2 = r2 % WW;
        const float* bp = beta + (size_t)(n2 * HP + h2) * WP + w2;
        float s = 0.f;
        #pragma unroll
        for (int kh = 0; kh < 3; ++kh)
            #pragma unroll
            for (int kw = 0; kw < 3; ++kw)
                s += bp[kh * WP + kw];
        kb[t] = s * (1.0f / 9.0f);
    }

    const v8f zero = {0.f, 0.f, 0.f, 0.f, 0.f, 0.f, 0.f, 0.f};
    v8f acc[2][4];
    #pragma unroll
    for (int mi = 0; mi < 2; ++mi)
        #pragma unroll
        for (int ni = 0; ni < 4; ++ni)
            acc[mi][ni] = zero;

    auto aOffOf = [&](int kt) -> int {
        const int k0   = kt << 7;
        const int pix  = k0 >> 8;
        const int half = (k0 >> 7) & 1;
        return ((pix / 3) * WP + (pix % 3)) * CIN + half * 128;
    };

    async_copy64(aDst0, aBase + aOffOf(0));
    async_copy64(bDst0, bBase);

    const int arow  = lane & 15;
    const int akoff = (lane >> 4) * 16;
    const int bkoff = (lane >> 4) * 16;

    for (int kt = 0; kt < 18; ++kt) {
        const int cur = kt & 1;

        wait_async0();
        __syncthreads();

        const uint8_t* Abuf = &lA[cur][0];
        const uint8_t* Bbuf = &lB[cur][0];

        v16i afr[2];
        #pragma unroll
        for (int mi = 0; mi < 2; ++mi) {
            const uint8_t* ap = Abuf + (wm * 32 + mi * 16 + arow) * 128 + akoff;
            #pragma unroll
            for (int j = 0; j < 8; ++j) {
                v2i c = *(const v2i*)(ap + (j >> 1) * 32 + (j & 1) * 8);
                afr[mi][2 * j]     = c[0];
                afr[mi][2 * j + 1] = c[1];
            }
        }
        v16i bfr[4];
        #pragma unroll
        for (int ni = 0; ni < 4; ++ni) {
            const uint8_t* bp2 = Bbuf + (wn * 64 + ni * 16 + (lane & 15)) * 128 + bkoff;
            #pragma unroll
            for (int j = 0; j < 4; ++j) {
                v4i c = *(const v4i*)(bp2 + j * 32);
                bfr[ni][4 * j]     = c[0];
                bfr[ni][4 * j + 1] = c[1];
                bfr[ni][4 * j + 2] = c[2];
                bfr[ni][4 * j + 3] = c[3];
            }
        }

        wait_ds0();
        __syncthreads();

        if (kt < 17) {
            async_copy64(cur ? aDst0 : aDst1, aBase + aOffOf(kt + 1));
            async_copy64(cur ? bDst0 : bDst1, bBase + ((kt + 1) << 7));
        }

        #pragma unroll
        for (int mi = 0; mi < 2; ++mi)
            #pragma unroll
            for (int ni = 0; ni < 4; ++ni)
                acc[mi][ni] = __builtin_amdgcn_wmma_f32_16x16x128_fp8_fp8(
                    afr[mi], bfr[ni], (short)0, acc[mi][ni], false, false);
    }

    const int colLocal = lane & 15;
    const int rowHalf  = (lane >> 4) * 8;
    #pragma unroll
    for (int ni = 0; ni < 4; ++ni) {
        const int co = n0 + wn * 64 + ni * 16 + colLocal;
        const float al = alpha[co];
        const float bi = bias[co];
        #pragma unroll
        for (int mi = 0; mi < 2; ++mi) {
            const int rbase = wm * 32 + mi * 16 + rowHalf;
            #pragma unroll
            for (int j = 0; j < 8; ++j) {
                const int r = rbase + j;
                const int m = m0 + r;
                out[(size_t)m * COUT + co] = acc[mi][ni][j] * (kb[r] * al) + bi;
            }
        }
    }
}

extern "C" void kernel_launch(void* const* d_in, const int* in_sizes, int n_in,
                              void* d_out, int out_size, void* d_ws, size_t ws_size,
                              hipStream_t stream) {
    (void)in_sizes; (void)n_in; (void)out_size; (void)ws_size;
    const float* x    = (const float*)d_in[0];
    const float* w    = (const float*)d_in[1];
    const float* bias = (const float*)d_in[2];
    float* out = (float*)d_out;

    uint8_t* ws = (uint8_t*)d_ws;
    size_t off = 0;
    uint8_t* binX  = ws + off; off += (size_t)NB * HP * WP * CIN;
    float*   beta  = (float*)(ws + off); off += (size_t)NB * HP * WP * sizeof(float);
    uint8_t* binKT = ws + off; off += (size_t)COUT * KDIM;
    off = (off + 255) & ~(size_t)255;
    float*   alpha = (float*)(ws + off); off += COUT * sizeof(float);

    prep_weights<<<COUT, 256, 0, stream>>>(w, binKT, alpha);
    prep_input<<<NB * HP * WP, 256, 0, stream>>>(x, binX, beta);

    dim3 grid(MTOT / 128, COUT / 128);
    bconv_gemm<<<grid, 256, 0, stream>>>(binX, binKT, beta, alpha, bias, out);
}
